// FocalWindowAttention_49349174231687
// MI455X (gfx1250) — compile-verified, matrix operand order checked by reading, not run on hardware
//
#include <hip/hip_runtime.h>
#include <hip/hip_bf16.h>

typedef __attribute__((ext_vector_type(16))) _Float16 v16h;
typedef __attribute__((ext_vector_type(8)))  _Float16 v8h;
typedef __attribute__((ext_vector_type(4)))  _Float16 v4h;
typedef __attribute__((ext_vector_type(8)))  float    v8f;

#define NH 12
#define HD 32
#define CDIM 384
#define WSZ 7
#define WS2 49
#define NKEY 132
#define NKTOT 181   // 49 + 132
#define NKPAD 192
#define SCALE_F 0.17677669529663689f

// -------------------------------------------------------------------------
// per-lane fragment load: 16 halves = two contiguous 16B chunks 16 halves apart
// matches CDNA5 16-bit A/B matrix VGPR layout (K = kb..kb+7, kb+16..kb+23)
__device__ __forceinline__ v16h ld_frag(const _Float16* p) {
    union { v16h v; v8h h[2]; } u;
    u.h[0] = *(const v8h*)(p);
    u.h[1] = *(const v8h*)(p + 16);
    return u.v;
}

// -------------------------------------------------------------------------
// init: rolled-neighbor gather offsets (132 entries), matches VALID_IND order
__global__ void nbr_init_kernel(int2* __restrict__ nbr) {
    if (threadIdx.x == 0 && blockIdx.x == 0) {
        const int sy[4] = { 3,  3, -3, -3 };   // source offset = -shift
        const int sx[4] = { 3, -3,  3, -3 };
        int kk = 0;
        for (int c = 0; c < 4; ++c)
            for (int i = 0; i < WSZ; ++i)
                for (int j = 0; j < WSZ; ++j) {
                    bool invalid;
                    if      (c == 0) invalid = (i < 4) && (j < 4);
                    else if (c == 1) invalid = (i < 4) && (j >= 3);
                    else if (c == 2) invalid = (i >= 3) && (j < 4);
                    else             invalid = (i >= 3) && (j >= 3);
                    if (!invalid) {
                        nbr[kk].x = i + sy[c];
                        nbr[kk].y = j + sx[c];
                        ++kk;
                    }
                }
    }
}

// -------------------------------------------------------------------------
// QKV GEMM: qkv[m,n] = sum_k x[m,k]*w_qkv[n,k] + b[n], split into f16 q/k/v planes
__global__ __launch_bounds__(128)
void qkv_gemm_kernel(const float* __restrict__ x, const float* __restrict__ w,
                     const float* __restrict__ bqkv,
                     _Float16* __restrict__ qh, _Float16* __restrict__ kh,
                     _Float16* __restrict__ vh) {
    __shared__ __align__(16) _Float16 At[64 * 32];
    __shared__ __align__(16) _Float16 Bt[64 * 32];
    const int mt = blockIdx.x, nt = blockIdx.y;
    const int tid = threadIdx.x;
    const int lane = tid & 31, wave = tid >> 5;
    const int wm = (wave >> 1) * 32;
    const int wn = (wave & 1) * 32;
    const int lrow = tid >> 1;
    const int lcol = (tid & 1) * 16;
    const int r  = lane & 15;
    const int kb = (lane >> 4) * 8;
    v8f acc[2][2] = {};

    for (int kt = 0; kt < CDIM; kt += 32) {
        const float4* ag = (const float4*)(x + (size_t)(mt * 64 + lrow) * CDIM + kt + lcol);
        const float4* bg = (const float4*)(w + (size_t)(nt * 64 + lrow) * CDIM + kt + lcol);
#pragma unroll
        for (int i = 0; i < 4; ++i) {
            float4 fa = ag[i];
            float4 fb = bg[i];
            v4h ha = { (_Float16)fa.x, (_Float16)fa.y, (_Float16)fa.z, (_Float16)fa.w };
            v4h hb = { (_Float16)fb.x, (_Float16)fb.y, (_Float16)fb.z, (_Float16)fb.w };
            *(v4h*)&At[lrow * 32 + lcol + i * 4] = ha;
            *(v4h*)&Bt[lrow * 32 + lcol + i * 4] = hb;
        }
        __syncthreads();
        v16h af[2], bf[2];
#pragma unroll
        for (int mi = 0; mi < 2; ++mi) af[mi] = ld_frag(&At[(wm + mi * 16 + r) * 32 + kb]);
#pragma unroll
        for (int ni = 0; ni < 2; ++ni) bf[ni] = ld_frag(&Bt[(wn + ni * 16 + r) * 32 + kb]);
#pragma unroll
        for (int mi = 0; mi < 2; ++mi)
#pragma unroll
            for (int ni = 0; ni < 2; ++ni)
                acc[mi][ni] = __builtin_amdgcn_wmma_f32_16x16x32_f16(
                    false, af[mi], false, bf[ni], (short)0, acc[mi][ni], false, false);
        __syncthreads();
    }

    const int hs = lane >> 4;
#pragma unroll
    for (int mi = 0; mi < 2; ++mi)
#pragma unroll
        for (int ni = 0; ni < 2; ++ni) {
            int n = nt * 64 + wn + ni * 16 + r;          // 0..1151
            _Float16* plane = (n < 384) ? qh : ((n < 768) ? kh : vh);
            int ch = (n < 384) ? n : ((n < 768) ? n - 384 : n - 768);
            float bias = bqkv[n];
#pragma unroll
            for (int rr = 0; rr < 8; ++rr) {
                int gm = mt * 64 + wm + mi * 16 + rr + 8 * hs;
                plane[(size_t)gm * CDIM + ch] = (_Float16)(acc[mi][ni][rr] + bias);
            }
        }
}

// -------------------------------------------------------------------------
// attention: one block per (window, head); 4 waves, each owns 16 query rows
__global__ __launch_bounds__(128)
void attn_kernel(const _Float16* __restrict__ qh, const _Float16* __restrict__ kh,
                 const _Float16* __restrict__ vh, const float* __restrict__ rpt,
                 const float* __restrict__ rpn, const int2* __restrict__ nbr,
                 _Float16* __restrict__ ao) {
    __shared__ __align__(16) _Float16 qs[64 * 32];        // padded queries
    __shared__ __align__(16) _Float16 ks[NKPAD * 32];     // keys (N x K layout)
    __shared__ __align__(16) _Float16 vts[HD * NKPAD];    // V transposed (dim x key)
    __shared__ __align__(16) _Float16 ps[64 * NKPAD];     // softmax probs f16

    const int bx = blockIdx.x;
    const int h = bx % NH, win = bx / NH;
    const int b0 = win / 576, wi = win % 576;
    const int wy = wi / 24, wx = wi % 24;
    const int tid = threadIdx.x, lane = tid & 31, wave = tid >> 5;

    // ---- cooperative gather of q / k_all / v_all (16B chunks) ----
    for (int t = tid; t < (64 + NKPAD + NKPAD) * 4; t += 128) {
        int ch = t & 3, rowid = t >> 2;
        int tensor, row;
        if (rowid < 64)       { tensor = 0; row = rowid; }
        else if (rowid < 256) { tensor = 1; row = rowid - 64; }
        else                  { tensor = 2; row = rowid - 256; }
        bool valid = (tensor == 0) ? (row < WS2) : (row < NKTOT);
        v8h data{};
        if (valid) {
            int y, x;
            if (row < WS2) {
                y = wy * 7 + row / 7;
                x = wx * 7 + row % 7;
            } else {
                int2 o = nbr[row - WS2];
                y = wy * 7 + o.x;
                x = wx * 7 + o.y;
                if (y < 0) y += 168; else if (y >= 168) y -= 168;
                if (x < 0) x += 168; else if (x >= 168) x -= 168;
            }
            const _Float16* plane = (tensor == 0) ? qh : ((tensor == 1) ? kh : vh);
            data = *(const v8h*)(plane + (((size_t)b0 * 168 + y) * 168 + x) * CDIM + h * HD + ch * 8);
        }
        if (tensor == 0)      *(v8h*)&qs[row * 32 + ch * 8] = data;
        else if (tensor == 1) *(v8h*)&ks[row * 32 + ch * 8] = data;
        else {
#pragma unroll
            for (int e = 0; e < 8; ++e) vts[(ch * 8 + e) * NKPAD + row] = data[e];
        }
    }
    __syncthreads();

    const int m0 = wave * 16;
    const int cn = lane & 15, hs = lane >> 4, kb = hs * 8;

    // ---- S = q . k^T (16 x 192, K=32) ----
    v16h aq = ld_frag(&qs[(m0 + cn) * 32 + kb]);
    v8f s[12];
#pragma unroll
    for (int ntl = 0; ntl < 12; ++ntl) {
        v16h bk = ld_frag(&ks[(ntl * 16 + cn) * 32 + kb]);
        v8f z = {};
        s[ntl] = __builtin_amdgcn_wmma_f32_16x16x32_f16(false, aq, false, bk, (short)0, z, false, false);
    }

    // ---- scale + bias + row max ----
    float rmax[8];
#pragma unroll
    for (int rr = 0; rr < 8; ++rr) rmax[rr] = -3.0e38f;
#pragma unroll
    for (int ntl = 0; ntl < 12; ++ntl) {
        int n = ntl * 16 + cn;
#pragma unroll
        for (int rr = 0; rr < 8; ++rr) {
            int qi = m0 + rr + 8 * hs;
            int qc = (qi > 48) ? 48 : qi;
            float v = s[ntl][rr] * SCALE_F;
            if (n < WS2) {
                int idx = ((qc / 7) - (n / 7) + 6) * 13 + ((qc % 7) - (n % 7) + 6);
                v += rpt[idx * NH + h];
            } else if (n < NKTOT) {
                v += rpn[((size_t)(h * WS2 + qc)) * NKEY + (n - WS2)];
            } else {
                v = -1.0e30f;
            }
            s[ntl][rr] = v;
            rmax[rr] = fmaxf(rmax[rr], v);
        }
    }
#pragma unroll
    for (int rr = 0; rr < 8; ++rr)
#pragma unroll
        for (int m = 1; m < 16; m <<= 1)
            rmax[rr] = fmaxf(rmax[rr], __shfl_xor(rmax[rr], m, 16));

    // ---- exp + row sum ----
    float rsum[8];
#pragma unroll
    for (int rr = 0; rr < 8; ++rr) rsum[rr] = 0.0f;
#pragma unroll
    for (int ntl = 0; ntl < 12; ++ntl)
#pragma unroll
        for (int rr = 0; rr < 8; ++rr) {
            float p = __expf(s[ntl][rr] - rmax[rr]);
            s[ntl][rr] = p;
            rsum[rr] += p;
        }
#pragma unroll
    for (int rr = 0; rr < 8; ++rr)
#pragma unroll
        for (int m = 1; m < 16; m <<= 1)
            rsum[rr] += __shfl_xor(rsum[rr], m, 16);
    float rinv[8];
#pragma unroll
    for (int rr = 0; rr < 8; ++rr) rinv[rr] = 1.0f / rsum[rr];

    // ---- stage P as f16 into this wave's private LDS region ----
#pragma unroll
    for (int ntl = 0; ntl < 12; ++ntl)
#pragma unroll
        for (int rr = 0; rr < 8; ++rr)
            ps[(m0 + rr + 8 * hs) * NKPAD + ntl * 16 + cn] =
                (_Float16)(s[ntl][rr] * rinv[rr]);

    // ---- O = P . V (16 x 32, K=192) ----
    v8f o[2] = {};
#pragma unroll
    for (int kst = 0; kst < 6; ++kst) {
        v16h ap = ld_frag(&ps[(m0 + cn) * NKPAD + kst * 32 + kb]);
#pragma unroll
        for (int ni = 0; ni < 2; ++ni) {
            v16h bv = ld_frag(&vts[(ni * 16 + cn) * NKPAD + kst * 32 + kb]);
            o[ni] = __builtin_amdgcn_wmma_f32_16x16x32_f16(false, ap, false, bv, (short)0, o[ni], false, false);
        }
    }

    // ---- store attention output (win,token,head-concat) ----
#pragma unroll
    for (int ni = 0; ni < 2; ++ni)
#pragma unroll
        for (int rr = 0; rr < 8; ++rr) {
            int qi = m0 + rr + 8 * hs;
            if (qi < WS2)
                ao[((size_t)win * WS2 + qi) * CDIM + h * HD + ni * 16 + cn] = (_Float16)o[ni][rr];
        }
}

// -------------------------------------------------------------------------
// proj GEMM: out[m,n] = sum_k ao[m,k]*w_proj[n,k] + b_proj[n]  (fp32 out)
__global__ __launch_bounds__(128)
void proj_gemm_kernel(const _Float16* __restrict__ a, const float* __restrict__ w,
                      const float* __restrict__ bias, float* __restrict__ out) {
    __shared__ __align__(16) _Float16 At[64 * 32];
    __shared__ __align__(16) _Float16 Bt[64 * 32];
    const int mt = blockIdx.x, nt = blockIdx.y;
    const int tid = threadIdx.x;
    const int lane = tid & 31, wave = tid >> 5;
    const int wm = (wave >> 1) * 32;
    const int wn = (wave & 1) * 32;
    const int lrow = tid >> 1;
    const int lcol = (tid & 1) * 16;
    const int r  = lane & 15;
    const int kb = (lane >> 4) * 8;
    v8f acc[2][2] = {};

    for (int kt = 0; kt < CDIM; kt += 32) {
        const v8h* ag = (const v8h*)(a + (size_t)(mt * 64 + lrow) * CDIM + kt + lcol);
        *(v8h*)&At[lrow * 32 + lcol]     = ag[0];
        *(v8h*)&At[lrow * 32 + lcol + 8] = ag[1];
        const float4* bg = (const float4*)(w + (size_t)(nt * 64 + lrow) * CDIM + kt + lcol);
#pragma unroll
        for (int i = 0; i < 4; ++i) {
            float4 fb = bg[i];
            v4h hb = { (_Float16)fb.x, (_Float16)fb.y, (_Float16)fb.z, (_Float16)fb.w };
            *(v4h*)&Bt[lrow * 32 + lcol + i * 4] = hb;
        }
        __syncthreads();
        v16h af[2], bf[2];
#pragma unroll
        for (int mi = 0; mi < 2; ++mi) af[mi] = ld_frag(&At[(wm + mi * 16 + r) * 32 + kb]);
#pragma unroll
        for (int ni = 0; ni < 2; ++ni) bf[ni] = ld_frag(&Bt[(wn + ni * 16 + r) * 32 + kb]);
#pragma unroll
        for (int mi = 0; mi < 2; ++mi)
#pragma unroll
            for (int ni = 0; ni < 2; ++ni)
                acc[mi][ni] = __builtin_amdgcn_wmma_f32_16x16x32_f16(
                    false, af[mi], false, bf[ni], (short)0, acc[mi][ni], false, false);
        __syncthreads();
    }

    const int hs = lane >> 4;
#pragma unroll
    for (int mi = 0; mi < 2; ++mi)
#pragma unroll
        for (int ni = 0; ni < 2; ++ni) {
            int n = nt * 64 + wn + ni * 16 + r;
            float b = bias[n];
#pragma unroll
            for (int rr = 0; rr < 8; ++rr) {
                int gm = mt * 64 + wm + mi * 16 + rr + 8 * hs;
                out[(size_t)gm * CDIM + n] = acc[mi][ni][rr] + b;
            }
        }
}

// -------------------------------------------------------------------------
extern "C" void kernel_launch(void* const* d_in, const int* in_sizes, int n_in,
                              void* d_out, int out_size, void* d_ws, size_t ws_size,
                              hipStream_t stream) {
    const float* x      = (const float*)d_in[0];
    const float* w_qkv  = (const float*)d_in[1];
    const float* b_qkv  = (const float*)d_in[2];
    const float* w_proj = (const float*)d_in[3];
    const float* b_proj = (const float*)d_in[4];
    const float* rpt    = (const float*)d_in[5];
    const float* rpn    = (const float*)d_in[6];
    // d_in[7]/d_in[8] are H,W (device-resident); setup fixes H=W=168.

    const int M  = in_sizes[0] / CDIM;   // B*H*W tokens (56448 for B=2)
    const int nW = M / WS2;              // total windows (1152)

    char* wsb = (char*)d_ws;
    int2* nbr = (int2*)wsb;
    const size_t SZ = (size_t)M * CDIM * sizeof(_Float16);
    _Float16* qh = (_Float16*)(wsb + 4096);
    _Float16* kh = (_Float16*)(wsb + 4096 + SZ);
    _Float16* vh = (_Float16*)(wsb + 4096 + 2 * SZ);
    _Float16* ao = (_Float16*)(wsb + 4096 + 3 * SZ);
    float* out = (float*)d_out;

    nbr_init_kernel<<<1, 32, 0, stream>>>(nbr);
    qkv_gemm_kernel<<<dim3(M / 64, (3 * CDIM) / 64), 128, 0, stream>>>(
        x, w_qkv, b_qkv, qh, kh, vh);
    attn_kernel<<<nW * NH, 128, 0, stream>>>(qh, kh, vh, rpt, rpn, nbr, ao);
    proj_gemm_kernel<<<dim3(M / 64, CDIM / 64), 128, 0, stream>>>(
        ao, w_proj, b_proj, out);
    (void)n_in; (void)out_size; (void)ws_size;
}
